// StageBGnnPolicy_45397804319393
// MI455X (gfx1250) — compile-verified, matrix operand order checked by reading, not run on hardware
//
#include <hip/hip_runtime.h>

typedef __attribute__((ext_vector_type(16))) _Float16 v16h;
typedef __attribute__((ext_vector_type(8)))  float    v8f;

#define DEVINL __device__ __forceinline__

constexpr int B = 16, C = 64, U = 640, S = 256, P = 273, E = 1024;
constexpr int CF = 64, UF = 48, EF = 16, H = 256;
constexpr int SPC = S / C;   // 4
constexpr int UPC = U / C;   // 10

#if defined(__AMDGCN__) && __has_builtin(__builtin_amdgcn_global_load_async_to_lds_b32) && __has_builtin(__builtin_amdgcn_s_wait_asynccnt)
#define USE_ASYNC_LDS 1
typedef __attribute__((address_space(1))) int  as1_int;
typedef __attribute__((address_space(3))) int  as3_int;
#else
#define USE_ASYNC_LDS 0
#endif

// Map (lane, element) -> k offset within a 32-wide K step for the 16-bit
// WMMA A fragment layout (ISA 7.12.2): lanes 0-15 rows M=0..15 hold
// K {0..7,16..23}; lanes 16-31 same rows hold K {8..15,24..31}.
DEVINL int kmap(int lane, int e) {
  int j = e >> 1, bit = e & 1;
  return ((lane >> 4) << 3) + ((j & 3) << 1) + bit + ((j >= 4) ? 16 : 0);
}

DEVINL v8f wmma_f16(v16h a, v16h b, v8f c) {
  return __builtin_amdgcn_wmma_f32_16x16x32_f16(false, a, false, b, (short)0, c,
                                                false, false);
}

// ---------------------------------------------------------------------------
// Pack a K x N (or transposed N x K) f32 matrix into f16 WMMA B-fragment
// records: layout [batch][kstep][ntile][lane][16 halves] (32B per lane).
// ---------------------------------------------------------------------------
__global__ __launch_bounds__(256) void pack_b_frag(
    const float* __restrict__ in, _Float16* __restrict__ out,
    int Klog, int Kpad, int N, int transposed,
    long in_bstride, long out_bstride, int nbatch) {
  int ksteps = Kpad >> 5, ntiles = N >> 4;
  long total = (long)nbatch * ksteps * ntiles * 32;
  long t = (long)blockIdx.x * 256 + threadIdx.x;
  if (t >= total) return;
  int lane = (int)(t & 31);
  long r = t >> 5;
  int ntile = (int)(r % ntiles); r /= ntiles;
  int kstep = (int)(r % ksteps); r /= ksteps;
  int batch = (int)r;
  const float* src = in + (long)batch * in_bstride;
  _Float16* dst = out + (long)batch * out_bstride +
                  (((long)kstep * ntiles + ntile) * 32 + lane) * 16;
  int n = ntile * 16 + (lane & 15);
#pragma unroll
  for (int e = 0; e < 16; e++) {
    int k = kstep * 32 + kmap(lane, e);
    float v = 0.f;
    if (k < Klog) v = transposed ? src[(long)n * Klog + k] : src[(long)k * N + n];
    dst[e] = (_Float16)v;
  }
}

// ---------------------------------------------------------------------------
// Generic WMMA GEMM: out = epilogue(A @ B), optional bias/relu/residual/LN.
// Block = 256 threads = 8 waves.  Each block computes a 32x256 output tile:
// each wave owns 2 row tiles x 2 col tiles (4 accumulators) so every
// B-fragment fetch feeds two v_wmma ops.  LN requires N==256 (one col group).
// ---------------------------------------------------------------------------
__global__ __launch_bounds__(256) void gemm_wmma(
    const float* __restrict__ A, int lda, int M, int Klog, int Kpad,
    const _Float16* __restrict__ Bfrag, long b_bstride, int rows_per_batch,
    int N, const float* __restrict__ bias, int relu, float scale,
    const float* __restrict__ residual, int ldres,
    const float* __restrict__ gamma, const float* __restrict__ beta,
    float* __restrict__ out, long ldout) {
  __shared__ __align__(32) _Float16 sA[2 * 768 * 16];
  __shared__ float sOut[32 * 260];
  __shared__ float sMean[32], sRstd[32];

  int tid = threadIdx.x, lane = tid & 31, w = tid >> 5;
  int rowbase = blockIdx.x * 32;
  int ksteps = Kpad >> 5, ntiles = N >> 4;
  int batch = rowbase / rows_per_batch;
  const _Float16* Bb = Bfrag + (long)batch * b_bstride;

  // Stage 32 x Kpad A rows into fragment order (two row tiles), f32 -> f16.
  for (int idx = tid; idx < ksteps * 64; idx += 256) {
    int ks = idx >> 6, rem = idx & 63, rt = rem >> 5, l2 = rem & 31;
    int row = rowbase + rt * 16 + (l2 & 15);
#pragma unroll
    for (int e = 0; e < 16; e++) {
      int k = ks * 32 + kmap(l2, e);
      float v = 0.f;
      if (row < M && k < Klog) v = A[(long)row * lda + k];
      sA[idx * 16 + e] = (_Float16)v;
    }
  }
  __syncthreads();

  int gt0 = blockIdx.y * 16 + w * 2;
  int gt1 = gt0 + 1;
  bool v0 = gt0 < ntiles, v1 = gt1 < ntiles;  // v1 implies v0
  v8f a00 = {}, a01 = {}, a10 = {}, a11 = {};
  if (v1) {
    for (int ks = 0; ks < ksteps; ks++) {
      const _Float16* bp0 = &Bb[(((long)ks * ntiles + gt0) * 32 + lane) * 16];
      const _Float16* bp1 = &Bb[(((long)ks * ntiles + gt1) * 32 + lane) * 16];
      if (ks + 1 < ksteps)
        __builtin_prefetch(&Bb[(((long)(ks + 1) * ntiles + gt0) * 32 + lane) * 16], 0, 1);
      v16h b0 = *(const v16h*)bp0;
      v16h b1 = *(const v16h*)bp1;
      v16h x0 = *(const v16h*)&sA[((long)ks * 64 + lane) * 16];
      v16h x1 = *(const v16h*)&sA[((long)ks * 64 + 32 + lane) * 16];
      a00 = wmma_f16(x0, b0, a00);
      a01 = wmma_f16(x0, b1, a01);
      a10 = wmma_f16(x1, b0, a10);
      a11 = wmma_f16(x1, b1, a11);
    }
  } else if (v0) {
    for (int ks = 0; ks < ksteps; ks++) {
      v16h b0 = *(const v16h*)&Bb[(((long)ks * ntiles + gt0) * 32 + lane) * 16];
      v16h x0 = *(const v16h*)&sA[((long)ks * 64 + lane) * 16];
      v16h x1 = *(const v16h*)&sA[((long)ks * 64 + 32 + lane) * 16];
      a00 = wmma_f16(x0, b0, a00);
      a10 = wmma_f16(x1, b0, a10);
    }
  }

  int rsel = lane >> 4, nl = lane & 15;
  if (gamma) {  // LN path (N==256, single col group; all 16 tiles valid)
#pragma unroll
    for (int r = 0; r < 8; r++) {
      int m0 = r + 8 * rsel, m1 = m0 + 16;
      int row0 = rowbase + m0, row1 = rowbase + m1;
      int c0 = gt0 * 16 + nl, c1 = gt1 * 16 + nl;
      float y00 = a00[r] * scale, y01 = a01[r] * scale;
      float y10 = a10[r] * scale, y11 = a11[r] * scale;
      if (bias) { y00 += bias[c0]; y01 += bias[c1]; y10 += bias[c0]; y11 += bias[c1]; }
      if (relu) {
        y00 = fmaxf(y00, 0.f); y01 = fmaxf(y01, 0.f);
        y10 = fmaxf(y10, 0.f); y11 = fmaxf(y11, 0.f);
      }
      if (residual) {
        if (row0 < M) {
          y00 += residual[(long)row0 * ldres + c0];
          y01 += residual[(long)row0 * ldres + c1];
        }
        if (row1 < M) {
          y10 += residual[(long)row1 * ldres + c0];
          y11 += residual[(long)row1 * ldres + c1];
        }
      }
      sOut[m0 * 260 + c0] = y00; sOut[m0 * 260 + c1] = y01;
      sOut[m1 * 260 + c0] = y10; sOut[m1 * 260 + c1] = y11;
    }
    __syncthreads();
    if (tid < 32) {
      float m = 0.f;
      for (int c2 = 0; c2 < 256; c2++) m += sOut[tid * 260 + c2];
      m *= (1.f / 256.f);
      float var = 0.f;
      for (int c2 = 0; c2 < 256; c2++) {
        float d = sOut[tid * 260 + c2] - m;
        var += d * d;
      }
      var *= (1.f / 256.f);
      sMean[tid] = m;
      sRstd[tid] = rsqrtf(var + 1e-5f);
    }
    __syncthreads();
    for (int idx = tid; idx < 32 * 256; idx += 256) {
      int r2 = idx >> 8, c2 = idx & 255;
      int row = rowbase + r2;
      if (row < M)
        out[(long)row * ldout + c2] =
            (sOut[r2 * 260 + c2] - sMean[r2]) * sRstd[r2] * gamma[c2] + beta[c2];
    }
  } else {  // direct store
#pragma unroll
    for (int r = 0; r < 8; r++) {
      int row0 = rowbase + r + 8 * rsel;
      int row1 = row0 + 16;
      if (v0) {
        int col = gt0 * 16 + nl;
        float y0 = a00[r] * scale, y1 = a10[r] * scale;
        if (bias) { y0 += bias[col]; y1 += bias[col]; }
        if (relu) { y0 = fmaxf(y0, 0.f); y1 = fmaxf(y1, 0.f); }
        if (residual) {
          if (row0 < M) y0 += residual[(long)row0 * ldres + col];
          if (row1 < M) y1 += residual[(long)row1 * ldres + col];
        }
        if (row0 < M) out[(long)row0 * ldout + col] = y0;
        if (row1 < M) out[(long)row1 * ldout + col] = y1;
      }
      if (v1) {
        int col = gt1 * 16 + nl;
        float y0 = a01[r] * scale, y1 = a11[r] * scale;
        if (bias) { y0 += bias[col]; y1 += bias[col]; }
        if (relu) { y0 = fmaxf(y0, 0.f); y1 = fmaxf(y1, 0.f); }
        if (residual) {
          if (row0 < M) y0 += residual[(long)row0 * ldres + col];
          if (row1 < M) y1 += residual[(long)row1 * ldres + col];
        }
        if (row0 < M) out[(long)row0 * ldout + col] = y0;
        if (row1 < M) out[(long)row1 * ldout + col] = y1;
      }
    }
  }
}

// ---------------------------------------------------------------------------
// Fused prg stage: per (b,c, p-tile of 32):
//   prg_ctx = LN(relu(cell_part[bc] + prg_part[p] + b))      (in LDS)
//   q       = prg_ctx @ Wq + qb                              (WMMA, LDS)
//   logits  = (q @ slot_k[b]^T) * scale  -> out cols 0..255
//   null    = prg_ctx . null_W + null_b  -> out col 256
// Never materializes the 286MB prg_ctx / prg_q tensors in HBM.
// ---------------------------------------------------------------------------
__global__ __launch_bounds__(256) void prg_fused(
    const float* __restrict__ cell_part, const float* __restrict__ prg_part,
    const float* __restrict__ pb, const float* __restrict__ pg,
    const float* __restrict__ pbb,
    const _Float16* __restrict__ wq, const float* __restrict__ qb,
    const _Float16* __restrict__ skf, long sk_bstride,
    const float* __restrict__ nw, const float* __restrict__ nb,
    float* __restrict__ out, float scale) {
  __shared__ float sX[32 * 260];
  __shared__ __align__(32) _Float16 sA[2 * 256 * 16];
  __shared__ float sMean[32], sRstd[32];

  int bc = blockIdx.x, b = bc / C;
  int ptile = blockIdx.y;
  int tid = threadIdx.x, lane = tid & 31, w = tid >> 5;

  // prg_pre + relu (32 rows)
  for (int idx = tid; idx < 32 * 256; idx += 256) {
    int r = idx >> 8, col = idx & 255;
    int p = ptile * 32 + r;
    float v = 0.f;
    if (p < P) v = cell_part[(long)bc * H + col] + prg_part[(long)p * H + col] + pb[col];
    sX[r * 260 + col] = fmaxf(v, 0.f);
  }
  __syncthreads();
  if (tid < 32) {
    float m = 0.f;
    for (int c2 = 0; c2 < 256; c2++) m += sX[tid * 260 + c2];
    m *= (1.f / 256.f);
    float var = 0.f;
    for (int c2 = 0; c2 < 256; c2++) {
      float d = sX[tid * 260 + c2] - m;
      var += d * d;
    }
    var *= (1.f / 256.f);
    sMean[tid] = m;
    sRstd[tid] = rsqrtf(var + 1e-5f);
  }
  __syncthreads();
  // stage LN'd prg_ctx as f16 A-fragments (two row tiles)
  for (int idx = tid; idx < 512; idx += 256) {
    int ks = idx >> 6, rem = idx & 63, rt = rem >> 5, l2 = rem & 31;
    int r = rt * 16 + (l2 & 15);
    float m = sMean[r], rs = sRstd[r];
#pragma unroll
    for (int e = 0; e < 16; e++) {
      int k = ks * 32 + kmap(l2, e);
      sA[idx * 16 + e] = (_Float16)((sX[r * 260 + k] - m) * rs * pg[k] + pbb[k]);
    }
  }
  // null head (32 rows, serial per thread over 256 cols)
  if (tid < 32) {
    int r = tid;
    float m = sMean[r], rs = sRstd[r], acc = 0.f;
    for (int c2 = 0; c2 < 256; c2++)
      acc += ((sX[r * 260 + c2] - m) * rs * pg[c2] + pbb[c2]) * nw[c2];
    int p = ptile * 32 + r;
    if (p < P) out[((long)bc * P + p) * 257 + 256] = acc + nb[0];
  }
  __syncthreads();

  int gt0 = w * 2, gt1 = gt0 + 1;
  // GEMM1: q = prg_ctx @ Wq
  v8f q00 = {}, q01 = {}, q10 = {}, q11 = {};
#pragma unroll
  for (int ks = 0; ks < 8; ks++) {
    v16h b0 = *(const v16h*)&wq[(((long)ks * 16 + gt0) * 32 + lane) * 16];
    v16h b1 = *(const v16h*)&wq[(((long)ks * 16 + gt1) * 32 + lane) * 16];
    v16h x0 = *(const v16h*)&sA[((long)ks * 64 + lane) * 16];
    v16h x1 = *(const v16h*)&sA[((long)ks * 64 + 32 + lane) * 16];
    q00 = wmma_f16(x0, b0, q00);
    q01 = wmma_f16(x0, b1, q01);
    q10 = wmma_f16(x1, b0, q10);
    q11 = wmma_f16(x1, b1, q11);
  }
  __syncthreads();
  {  // q (+bias) -> sX
    int rsel = lane >> 4, nl = lane & 15;
    int c0 = gt0 * 16 + nl, c1 = gt1 * 16 + nl;
#pragma unroll
    for (int r = 0; r < 8; r++) {
      int m0 = r + 8 * rsel, m1 = m0 + 16;
      sX[m0 * 260 + c0] = q00[r] + qb[c0];
      sX[m0 * 260 + c1] = q01[r] + qb[c1];
      sX[m1 * 260 + c0] = q10[r] + qb[c0];
      sX[m1 * 260 + c1] = q11[r] + qb[c1];
    }
  }
  __syncthreads();
  // re-stage q as A fragments
  for (int idx = tid; idx < 512; idx += 256) {
    int ks = idx >> 6, rem = idx & 63, rt = rem >> 5, l2 = rem & 31;
    int r = rt * 16 + (l2 & 15);
#pragma unroll
    for (int e = 0; e < 16; e++) {
      int k = ks * 32 + kmap(l2, e);
      sA[idx * 16 + e] = (_Float16)sX[r * 260 + k];
    }
  }
  __syncthreads();
  // GEMM2: logits = q @ slot_k[b]^T
  const _Float16* bk = skf + (long)b * sk_bstride;
  v8f l00 = {}, l01 = {}, l10 = {}, l11 = {};
#pragma unroll
  for (int ks = 0; ks < 8; ks++) {
    v16h b0 = *(const v16h*)&bk[(((long)ks * 16 + gt0) * 32 + lane) * 16];
    v16h b1 = *(const v16h*)&bk[(((long)ks * 16 + gt1) * 32 + lane) * 16];
    v16h x0 = *(const v16h*)&sA[((long)ks * 64 + lane) * 16];
    v16h x1 = *(const v16h*)&sA[((long)ks * 64 + 32 + lane) * 16];
    l00 = wmma_f16(x0, b0, l00);
    l01 = wmma_f16(x0, b1, l01);
    l10 = wmma_f16(x1, b0, l10);
    l11 = wmma_f16(x1, b1, l11);
  }
  {
    int rsel = lane >> 4, nl = lane & 15;
    int c0 = gt0 * 16 + nl, c1 = gt1 * 16 + nl;
#pragma unroll
    for (int r = 0; r < 8; r++) {
      int m0 = r + 8 * rsel, m1 = m0 + 16;
      int p0 = ptile * 32 + m0, p1 = ptile * 32 + m1;
      if (p0 < P) {
        long base = ((long)bc * P + p0) * 257;
        out[base + c0] = l00[r] * scale;
        out[base + c1] = l01[r] * scale;
      }
      if (p1 < P) {
        long base = ((long)bc * P + p1) * 257;
        out[base + c0] = l10[r] * scale;
        out[base + c1] = l11[r] * scale;
      }
    }
  }
}

// ---------------------------------------------------------------------------
// Elementwise gather / concat kernels.
// ---------------------------------------------------------------------------
__global__ __launch_bounds__(256) void build_catE(
    const float* __restrict__ cell_h, const int* __restrict__ eidx,
    const float* __restrict__ eattr, float* __restrict__ out) {
  long t = (long)blockIdx.x * 256 + threadIdx.x;
  long total = (long)B * E * 288;
  if (t >= total) return;
  int col = (int)(t % 288);
  long be = t / 288;
  int b = (int)(be / E);
  float v = 0.f;
  if (col < 256) {
    int s = eidx[be * 2 + 0];
    s = s < 0 ? 0 : (s > C - 1 ? C - 1 : s);
    v = cell_h[((long)b * C + s) * H + col];
  } else if (col < 272) {
    v = eattr[be * EF + (col - 256)];
  }
  out[t] = v;
}

__global__ __launch_bounds__(256) void concat2_256(
    const float* __restrict__ x0, const float* __restrict__ x1,
    float* __restrict__ out, long rows) {
  long t = (long)blockIdx.x * 256 + threadIdx.x;
  if (t >= rows * 512) return;
  int col = (int)(t % 512);
  long r = t / 512;
  out[t] = (col < 256) ? x0[r * 256 + col] : x1[r * 256 + (col - 256)];
}

__global__ __launch_bounds__(256) void build_catU(
    const float* __restrict__ ue_h0, const float* __restrict__ cell_h,
    float* __restrict__ out) {
  long t = (long)blockIdx.x * 256 + threadIdx.x;
  if (t >= (long)B * U * 512) return;
  int col = (int)(t % 512);
  long bu = t / 512;
  int u = (int)(bu % U), b = (int)(bu / U);
  out[t] = (col < 256) ? ue_h0[bu * 256 + col]
                       : cell_h[((long)b * C + u / UPC) * H + (col - 256)];
}

__global__ __launch_bounds__(256) void build_catS(
    const float* __restrict__ cell_h, const float* __restrict__ spos,
    const float* __restrict__ slpos, float* __restrict__ out) {
  long t = (long)blockIdx.x * 256 + threadIdx.x;
  if (t >= (long)B * S * 768) return;
  int col = (int)(t % 768);
  long bs = t / 768;
  int s = (int)(bs % S), b = (int)(bs / S);
  float v;
  if (col < 256)       v = cell_h[((long)b * C + s / SPC) * H + col];
  else if (col < 512)  v = spos[(long)s * H + (col - 256)];
  else                 v = slpos[(long)(s % SPC) * H + (col - 512)];
  out[t] = v;
}

// Deterministic segment mean: one block per (b,c), dst indices staged in LDS
// via the CDNA5 async global->LDS path when available (ASYNCcnt).
__global__ __launch_bounds__(256) void seg_mean(
    const float* __restrict__ msg, const int* __restrict__ eidx,
    float* __restrict__ agg) {
  __shared__ int sdst[E];
  int bc = blockIdx.x, b = bc / C, c = bc % C;
  int tid = threadIdx.x;
#if USE_ASYNC_LDS
  {
    for (int e = tid; e < E; e += 256)
      __builtin_amdgcn_global_load_async_to_lds_b32(
          (as1_int*)(eidx + ((long)b * E + e) * 2 + 1),
          (as3_int*)&sdst[e], 0, 0);
    __builtin_amdgcn_s_wait_asynccnt(0);
  }
  __syncthreads();
  for (int e = tid; e < E; e += 256) {
    int d = sdst[e];
    sdst[e] = d < 0 ? 0 : (d > C - 1 ? C - 1 : d);
  }
#else
  for (int e = tid; e < E; e += 256) {
    int d = eidx[((long)b * E + e) * 2 + 1];
    sdst[e] = d < 0 ? 0 : (d > C - 1 ? C - 1 : d);
  }
#endif
  __syncthreads();
  float acc = 0.f;
  int cnt = 0;
  for (int e = 0; e < E; e++) {
    if (sdst[e] == c) {
      acc += msg[((long)b * E + e) * H + tid];
      cnt++;
    }
  }
  agg[(long)bc * H + tid] = acc / fmaxf((float)cnt, 1.f);
}

// Null heads: one wave per row, dot over H=256 with shuffle reduce.
__global__ __launch_bounds__(256) void dot_head(
    const float* __restrict__ X, int ldx, int M,
    const float* __restrict__ wv, const float* __restrict__ bv,
    float* __restrict__ out, long ldout, long coloff) {
  int wid = threadIdx.x >> 5, lane = threadIdx.x & 31;
  int row = blockIdx.x * 8 + wid;
  if (row >= M) return;
  const float* x = X + (long)row * ldx;
  float acc = 0.f;
#pragma unroll
  for (int j = 0; j < 8; j++) {
    int k = lane + j * 32;
    acc += x[k] * wv[k];
  }
  for (int off = 16; off; off >>= 1) acc += __shfl_down(acc, off);
  if (lane == 0) out[(long)row * ldout + coloff] = acc + bv[0];
}

// Softmax over U=640 columns, one wave per row.
__global__ __launch_bounds__(256) void softmax_rows(
    const float* __restrict__ in, long ldin, float* __restrict__ out, int M) {
  constexpr int J = U / 32;  // 20
  int wid = threadIdx.x >> 5, lane = threadIdx.x & 31;
  int row = blockIdx.x * 8 + wid;
  if (row >= M) return;
  const float* src = in + (long)row * ldin;
  float v[J];
  float mx = -3.4e38f;
#pragma unroll
  for (int j = 0; j < J; j++) {
    v[j] = src[lane + j * 32];
    mx = fmaxf(mx, v[j]);
  }
  for (int off = 16; off; off >>= 1) mx = fmaxf(mx, __shfl_down(mx, off));
  mx = __shfl(mx, 0);
  float s = 0.f;
#pragma unroll
  for (int j = 0; j < J; j++) {
    v[j] = __expf(v[j] - mx);
    s += v[j];
  }
  for (int off = 16; off; off >>= 1) s += __shfl_down(s, off);
  s = __shfl(s, 0);
  float inv = 1.f / s;
#pragma unroll
  for (int j = 0; j < J; j++) out[(long)row * U + lane + j * 32] = v[j] * inv;
}

// ---------------------------------------------------------------------------
extern "C" void kernel_launch(void* const* d_in, const int* in_sizes, int n_in,
                              void* d_out, int out_size, void* d_ws,
                              size_t ws_size, hipStream_t stream) {
  (void)in_sizes; (void)n_in; (void)out_size; (void)ws_size;

  const float* cellF = (const float*)d_in[0];
  const float* ueF   = (const float*)d_in[1];
  const int*   eidx  = (const int*)d_in[2];
  const float* eattr = (const float*)d_in[3];
  auto PF = [&](int i) { return (const float*)d_in[i]; };

  char* base = (char*)d_ws;
  size_t off = 0;
  auto allocF = [&](size_t n) -> float* {
    float* p = (float*)(base + off);
    off += ((n * 4 + 255) / 256) * 256;
    return p;
  };
  auto allocH = [&](size_t n) -> _Float16* {
    _Float16* p = (_Float16*)(base + off);
    off += ((n * 2 + 255) / 256) * 256;
    return p;
  };

  float* cellA = allocF((size_t)B * C * H);
  float* cellB = allocF((size_t)B * C * H);
  float* catE  = allocF((size_t)B * E * 288);
  float* msgh  = allocF((size_t)B * E * H);
  float* msg2o = allocF((size_t)B * E * H);
  float* agg   = allocF((size_t)B * C * H);
  float* cat2  = allocF((size_t)B * C * 2 * H);
  float* updh  = allocF((size_t)B * C * H);
  float* ueh0  = allocF((size_t)B * U * H);
  float* catUb = allocF((size_t)B * U * 2 * H);
  float* ueh   = allocF((size_t)B * U * H);
  float* catSb = allocF((size_t)B * S * 3 * H);
  float* slotc = allocF((size_t)B * S * H);
  float* slotq = allocF((size_t)B * S * H);
  float* uek   = allocF((size_t)B * U * H);
  float* prob  = allocF((size_t)B * S * U);
  float* softc = allocF((size_t)B * S * H);
  float* catSU = allocF((size_t)B * S * 2 * H);
  float* slotp = allocF((size_t)B * S * H);
  float* slotk = allocF((size_t)B * S * H);
  float* cpart = allocF((size_t)B * C * H);
  float* ppart = allocF((size_t)288 * H);

  auto pack = [&](const float* Wp, int Klog, int Kpad, int N, int transposed,
                  int nbatch, long in_bstride) -> _Float16* {
    long per = (long)Kpad * N;
    _Float16* dst = allocH((size_t)per * nbatch);
    long total = (long)nbatch * (Kpad / 32) * (N / 16) * 32;
    pack_b_frag<<<(int)((total + 255) / 256), 256, 0, stream>>>(
        Wp, dst, Klog, Kpad, N, transposed, in_bstride, per, nbatch);
    return dst;
  };
  auto gemm = [&](const float* A, int lda, int M, int Klog, int Kpad,
                  const _Float16* Bf, long bstride, int rpb, int N,
                  const float* bias, int relu, float scale,
                  const float* residual, int ldres, const float* g,
                  const float* bb, float* out, long ldout) {
    dim3 grid((M + 31) / 32, (N + 255) / 256);
    gemm_wmma<<<grid, 256, 0, stream>>>(A, lda, M, Klog, Kpad, Bf, bstride, rpb,
                                        N, bias, relu, scale, residual, ldres,
                                        g, bb, out, ldout);
  };

  const float scale = 0.0625f;  // 1/sqrt(H)

  // cell_in: lrn
  _Float16* p_ci = pack(PF(4), CF, 64, H, 0, 1, 0);
  gemm(cellF, CF, B * C, CF, 64, p_ci, 0, B * C, H, PF(5), 1, 1.f, nullptr, 0,
       PF(6), PF(7), cellA, H);

  // 2 rounds of message passing
  float* cur = cellA;
  float* nxt = cellB;
  for (int i = 0; i < 2; i++) {
    int pb0 = 16 + i * 10;
    {
      long total = (long)B * E * 288;
      build_catE<<<(int)((total + 255) / 256), 256, 0, stream>>>(cur, eidx,
                                                                 eattr, catE);
    }
    _Float16* pm1 = pack(PF(pb0 + 0), H + EF, 288, H, 0, 1, 0);
    gemm(catE, 288, B * E, 288, 288, pm1, 0, B * E, H, PF(pb0 + 1), 1, 1.f,
         nullptr, 0, nullptr, nullptr, msgh, H);
    _Float16* pm2 = pack(PF(pb0 + 2), H, H, H, 0, 1, 0);
    gemm(msgh, H, B * E, H, H, pm2, 0, B * E, H, PF(pb0 + 3), 0, 1.f, nullptr,
         0, nullptr, nullptr, msg2o, H);
    seg_mean<<<B * C, 256, 0, stream>>>(msg2o, eidx, agg);
    {
      long total = (long)B * C * 512;
      concat2_256<<<(int)((total + 255) / 256), 256, 0, stream>>>(
          cur, agg, cat2, (long)B * C);
    }
    _Float16* pu1 = pack(PF(pb0 + 4), 2 * H, 512, H, 0, 1, 0);
    gemm(cat2, 512, B * C, 512, 512, pu1, 0, B * C, H, PF(pb0 + 5), 1, 1.f,
         nullptr, 0, nullptr, nullptr, updh, H);
    _Float16* pu2 = pack(PF(pb0 + 6), H, H, H, 0, 1, 0);
    gemm(updh, H, B * C, H, H, pu2, 0, B * C, H, PF(pb0 + 7), 0, 1.f, cur, H,
         PF(pb0 + 8), PF(pb0 + 9), nxt, H);
    float* t = cur; cur = nxt; nxt = t;
  }

  // ue_in lrn, fuse lrn
  _Float16* p_ui = pack(PF(8), UF, 64, H, 0, 1, 0);
  gemm(ueF, UF, B * U, UF, 64, p_ui, 0, B * U, H, PF(9), 1, 1.f, nullptr, 0,
       PF(10), PF(11), ueh0, H);
  {
    long total = (long)B * U * 512;
    build_catU<<<(int)((total + 255) / 256), 256, 0, stream>>>(ueh0, cur, catUb);
  }
  _Float16* p_fu = pack(PF(12), 2 * H, 512, H, 0, 1, 0);
  gemm(catUb, 512, B * U, 512, 512, p_fu, 0, B * U, H, PF(13), 1, 1.f, nullptr,
       0, PF(14), PF(15), ueh, H);

  // slot ctx
  {
    long total = (long)B * S * 768;
    build_catS<<<(int)((total + 255) / 256), 256, 0, stream>>>(cur, PF(42),
                                                               PF(43), catSb);
  }
  _Float16* p_sf = pack(PF(44), 3 * H, 768, H, 0, 1, 0);
  gemm(catSb, 768, B * S, 768, 768, p_sf, 0, B * S, H, PF(45), 1, 1.f, nullptr,
       0, PF(46), PF(47), slotc, H);

  _Float16* p_sq = pack(PF(36), H, H, H, 0, 1, 0);
  gemm(slotc, H, B * S, H, H, p_sq, 0, B * S, H, PF(37), 0, 1.f, nullptr, 0,
       nullptr, nullptr, slotq, H);
  _Float16* p_uk = pack(PF(38), H, H, H, 0, 1, 0);
  gemm(ueh, H, B * U, H, H, p_uk, 0, B * U, H, PF(39), 0, 1.f, nullptr, 0,
       nullptr, nullptr, uek, H);

  float* outUE = (float*)d_out;
  float* outPRG = outUE + (long)B * S * (U + 1);

  // ue attention: slot_q @ ue_k^T * scale  (batched B-fragments)
  _Float16* p_ukT = pack(uek, H, H, U, 1, B, (long)U * H);
  gemm(slotq, H, B * S, H, H, p_ukT, (long)H * U, S, U, nullptr, 0, scale,
       nullptr, 0, nullptr, nullptr, outUE, U + 1);
  dot_head<<<(B * S + 7) / 8, 256, 0, stream>>>(slotc, H, B * S, PF(40), PF(41),
                                                outUE, U + 1, U);
  softmax_rows<<<(B * S + 7) / 8, 256, 0, stream>>>(outUE, U + 1, prob, B * S);

  // soft ctx: prob @ ue_h  (K = U = 640)
  _Float16* p_uh = pack(ueh, U, U, H, 0, B, (long)U * H);
  gemm(prob, U, B * S, U, U, p_uh, (long)U * H, S, H, nullptr, 0, 1.f, nullptr,
       0, nullptr, nullptr, softc, H);
  {
    long total = (long)B * S * 512;
    concat2_256<<<(int)((total + 255) / 256), 256, 0, stream>>>(
        slotc, softc, catSU, (long)B * S);
  }
  _Float16* p_su = pack(PF(48), 2 * H, 512, H, 0, 1, 0);
  gemm(catSU, 512, B * S, 512, 512, p_su, 0, B * S, H, PF(49), 1, 1.f, nullptr,
       0, PF(50), PF(51), slotp, H);
  _Float16* p_pk = pack(PF(59), H, H, H, 0, 1, 0);
  gemm(slotp, H, B * S, H, H, p_pk, 0, B * S, H, PF(60), 0, 1.f, nullptr, 0,
       nullptr, nullptr, slotk, H);
  _Float16* p_skT = pack(slotk, H, H, S, 1, B, (long)S * H);

  // prg parts
  _Float16* p_wcc = pack(PF(53), H, H, H, 0, 1, 0);
  gemm(cur, H, B * C, H, H, p_wcc, 0, B * C, H, nullptr, 0, 1.f, nullptr, 0,
       nullptr, nullptr, cpart, H);
  _Float16* p_wcp = pack(PF(53) + (long)H * H, H, H, H, 0, 1, 0);
  gemm(PF(52), H, P, H, H, p_wcp, 0, P, H, nullptr, 0, 1.f, nullptr, 0, nullptr,
       nullptr, ppart, H);
  _Float16* p_wq = pack(PF(57), H, H, H, 0, 1, 0);

  // fused prg_ctx -> prg_q -> prg_logits (+null)
  prg_fused<<<dim3(B * C, (P + 31) / 32), 256, 0, stream>>>(
      cpart, ppart, PF(54), PF(55), PF(56), p_wq, PF(58), p_skT, (long)H * S,
      PF(61), PF(62), outPRG, scale);
}
